// TFM_TOKEN_Classifier_89309549953254
// MI455X (gfx1250) — matrix operand order repaired and checked by reading, not compiled, not run on hardware
//
#include <hip/hip_runtime.h>
#include <hip/hip_bf16.h>
#include <math.h>

// ---------------- problem constants (match reference) ----------------
#define BB   16
#define CC   16
#define TT   128
#define DD   256
#define HH   8
#define DH   32
#define LL   12
#define NCLS 5
#define FFD  1024
#define NT   2049      // C*T + 1 (cls token prepended)
#define EPSV 1e-5f

typedef __bf16 bf16_t;
typedef bf16_t v16bf __attribute__((ext_vector_type(16)));
typedef bf16_t v8bf  __attribute__((ext_vector_type(8)));
typedef float  v8f   __attribute__((ext_vector_type(8)));

// ---------------- CDNA5 async global->LDS helpers ----------------
// GLOBAL_LOAD_ASYNC_TO_LDS_B128: per-lane 16B DMA global->LDS, tracked by
// ASYNCcnt (no VGPR round-trip, no LOADcnt/DScnt serialization).
__device__ __forceinline__ void async_copy16(const bf16_t* g, bf16_t* lds) {
  unsigned lds_off = (unsigned)(unsigned long long)lds;  // flat LDS addr[31:0]
  asm volatile("global_load_async_to_lds_b128 %0, %1, off"
               :: "v"(lds_off), "v"(g)
               : "memory");
}
__device__ __forceinline__ void wait_async0() {
  asm volatile("s_wait_asynccnt 0x0" ::: "memory");
}

// ---------------- f32 -> bf16 weight conversion ----------------
__global__ void cvt_bf16_kernel(const float* __restrict__ src,
                                bf16_t* __restrict__ dst, size_t n) {
  size_t i = (size_t)blockIdx.x * blockDim.x + threadIdx.x;
  if (i < n) dst[i] = (bf16_t)src[i];
}

// ---------------- embedding + channel emb + sinusoidal PE ----------------
__global__ __launch_bounds__(DD) void embed_kernel(const int* __restrict__ x,
                                                   const float* __restrict__ emb,
                                                   const float* __restrict__ cht,
                                                   const float* __restrict__ cls,
                                                   float* __restrict__ h) {
  int n = blockIdx.x, b = blockIdx.y, d = threadIdx.x;
  float v;
  if (n == 0) {
    v = cls[d];
  } else {
    int c = (n - 1) / TT, t = (n - 1) % TT;
    int tok = x[(b * CC + c) * TT + t];
    int i2 = d & ~1;
    // div = exp(-(2i) * ln(10000)/D) ; ln(10000)/256 = 0.0359857045...
    float div = __expf(-(float)i2 * 0.0359857045f);
    float ang = (float)t * div;
    float pe = (d & 1) ? __cosf(ang) : __sinf(ang);
    v = emb[(size_t)tok * DD + d] + cht[c * DD + d] + pe;
  }
  h[((size_t)b * NT + n) * DD + d] = v;
}

// ---------------- LayerNorm (f32 in) -> bf16 out ----------------
__global__ __launch_bounds__(DD) void ln_kernel(const float* __restrict__ h,
                                                const float* __restrict__ g,
                                                const float* __restrict__ beta,
                                                bf16_t* __restrict__ y) {
  __shared__ float red[DD];
  int n = blockIdx.x, b = blockIdx.y, d = threadIdx.x;
  size_t idx = ((size_t)b * NT + n) * DD + d;
  float xv = h[idx];
  red[d] = xv; __syncthreads();
  for (int s = DD / 2; s > 0; s >>= 1) { if (d < s) red[d] += red[d + s]; __syncthreads(); }
  float m = red[0] * (1.0f / DD); __syncthreads();
  float dx = xv - m;
  red[d] = dx * dx; __syncthreads();
  for (int s = DD / 2; s > 0; s >>= 1) { if (d < s) red[d] += red[d + s]; __syncthreads(); }
  float var = red[0] * (1.0f / DD);
  y[idx] = (bf16_t)(dx * rsqrtf(var + EPSV) * g[d] + beta[d]);
}

// ---------------- WMMA bf16 GEMM: Out[b] = act(A[b] @ W + bias) (+Out) -------
// A: bf16 [batch][M][K] row-major.  W: bf16 [K][N] row-major (shared).
// Block: 256 thr = 8 waves, tile 64(M) x 64(N), K-step 32.
// Double-buffered LDS staged with GLOBAL_LOAD_ASYNC_TO_LDS_B128: tile k+1
// streams in while WMMAs consume tile k; one barrier + one asynccnt wait
// per K-step. A-row index is clamped (not predicated) so EXEC stays all-1s
// (WMMA requirement) and async lanes never skip.
template <int HAS_BIAS, int RES, int ACT_GELU, int OUT_BF16>
__global__ __launch_bounds__(256) void gemm_ws_kernel(
    const bf16_t* __restrict__ A, const bf16_t* __restrict__ W,
    const float* __restrict__ bias, void* __restrict__ Out,
    int M, int N, int K) {
  constexpr int LAS = 56;  // lds A row stride (bf16): 112B = 7*16B, conflict-spread
  constexpr int LBS = 72;  // lds W row stride (bf16): 144B = 9*16B
  __shared__ __align__(16) bf16_t lA[2][64 * LAS];
  __shared__ __align__(16) bf16_t lB[2][32 * LBS];

  const int b = blockIdx.z;
  const int m0 = blockIdx.y * 64, n0 = blockIdx.x * 64;
  const bf16_t* Ab = A + (size_t)b * M * K;

  const int tid = threadIdx.x;
  const int wave = tid >> 5, lane = tid & 31;
  const int half = lane >> 4, lm = lane & 15;
  const int mw = wave & 3, nw = wave >> 2;           // mw:0..3, nw:0..1
  const int asr = tid >> 2, asc = (tid & 3) * 8;     // A stage: 64 rows x 32 cols
  const int bsr = tid >> 3, bsc = (tid & 7) * 8;     // W stage: 32 rows x 64 cols

  // Clamped A row: always a valid address; epilogue guard discards extra rows.
  int gmA = m0 + asr; if (gmA > M - 1) gmA = M - 1;
  const bf16_t* agp = Ab + (size_t)gmA * K + asc;          // + k0
  const bf16_t* bgp = W + (size_t)bsr * N + (n0 + bsc);    // + k0*N

  v8f acc[2] = {};

  const int nk = K >> 5;  // K-steps of 32
  // prologue: stage tile 0 into buffer 0
  async_copy16(agp, &lA[0][asr * LAS + asc]);
  async_copy16(bgp, &lB[0][bsr * LBS + bsc]);

  for (int i = 0; i < nk; ++i) {
    const int buf = i & 1;
    wait_async0();      // my writes of tile i are in LDS
    __syncthreads();    // everyone's writes visible; prior reads of buf^1 retired
    if (i + 1 < nk) {   // stage tile i+1 into the other buffer (fully async)
      int k1 = (i + 1) << 5;
      async_copy16(agp + k1, &lA[buf ^ 1][asr * LAS + asc]);
      async_copy16(bgp + (size_t)k1 * N, &lB[buf ^ 1][bsr * LBS + bsc]);
    }

    // ---- A fragment: lane -> M=lm ; element e -> K = (e<8?e:e+8) + 8*half ----
    const int arow = mw * 16 + lm;
    v8bf alo = *(const v8bf*)(&lA[buf][arow * LAS + half * 8]);
    v8bf ahi = *(const v8bf*)(&lA[buf][arow * LAS + 16 + half * 8]);
    v16bf afrag = __builtin_shufflevector(alo, ahi, 0, 1, 2, 3, 4, 5, 6, 7,
                                          8, 9, 10, 11, 12, 13, 14, 15);
#pragma unroll
    for (int j = 0; j < 2; ++j) {
      // ---- B fragment: lane -> N=lm ; element e -> K = (e<8?e:e+8) + 8*half ----
      const bf16_t* bp = &lB[buf][nw * 32 + j * 16 + lm];
      v16bf bfrag;
#pragma unroll
      for (int e = 0; e < 16; ++e)
        bfrag[e] = bp[((e < 8 ? e : e + 8) + 8 * half) * LBS];
      acc[j] = __builtin_amdgcn_wmma_f32_16x16x32_bf16(
          false, afrag, false, bfrag, (short)0, acc[j], false, false);
    }
  }

  // ---- epilogue: D layout lane->N (lm), VGPR r -> M = r + 8*half ----
#pragma unroll
  for (int j = 0; j < 2; ++j) {
    int gn = n0 + nw * 32 + j * 16 + lm;
#pragma unroll
    for (int r = 0; r < 8; ++r) {
      int gm = m0 + mw * 16 + half * 8 + r;
      if (gm < M) {
        float v = acc[j][r];
        if (HAS_BIAS) v += bias[gn];
        if (ACT_GELU) v = 0.5f * v * (1.0f + erff(v * 0.70710678f));  // exact GELU
        size_t oi = ((size_t)b * M + gm) * N + gn;
        if (OUT_BF16) {
          ((bf16_t*)Out)[oi] = (bf16_t)v;
        } else {
          float* op = (float*)Out;
          float o = RES ? op[oi] : 0.0f;
          op[oi] = o + v;
        }
      }
    }
  }
}

// ---------------- q: softmax over 32 features, then * dh^-0.5 ----------------
__global__ void qsoftmax_kernel(float* __restrict__ q) {
  int idx = blockIdx.x * blockDim.x + threadIdx.x;
  const int total = BB * NT * HH;
  if (idx >= total) return;
  int hh = idx % HH;
  int n = (idx / HH) % NT;
  int b = idx / (HH * NT);
  float* p = q + ((size_t)b * NT + n) * DD + hh * DH;
  float m = -1e30f;
#pragma unroll
  for (int i = 0; i < DH; ++i) m = fmaxf(m, p[i]);
  float e[DH];
  float s = 0.f;
#pragma unroll
  for (int i = 0; i < DH; ++i) { e[i] = __expf(p[i] - m); s += e[i]; }
  float inv = 0.17677669529663687f / s;  // 32^-0.5 / sum
#pragma unroll
  for (int i = 0; i < DH; ++i) p[i] = e[i] * inv;
}

// ---------------- k: softmax over the sequence dim (2049) per (b,d) ----------
__global__ __launch_bounds__(256) void ksoftmax_kernel(float* __restrict__ k) {
  __shared__ float red[256];
  int d = blockIdx.x, b = blockIdx.y, t = threadIdx.x;
  float* base = k + (size_t)b * NT * DD + d;
  float m = -1e30f;
  for (int n = t; n < NT; n += 256) m = fmaxf(m, base[(size_t)n * DD]);
  red[t] = m; __syncthreads();
  for (int s = 128; s > 0; s >>= 1) { if (t < s) red[t] = fmaxf(red[t], red[t + s]); __syncthreads(); }
  m = red[0]; __syncthreads();
  float sum = 0.f;
  for (int n = t; n < NT; n += 256) sum += __expf(base[(size_t)n * DD] - m);
  red[t] = sum; __syncthreads();
  for (int s = 128; s > 0; s >>= 1) { if (t < s) red[t] += red[t + s]; __syncthreads(); }
  float inv = 1.0f / red[0];
  for (int n = t; n < NT; n += 256) {
    size_t o = (size_t)n * DD;
    base[o] = __expf(base[o] - m) * inv;
  }
}

// ---------------- ctx[b,h,d,e] = sum_n k[b,n,h,d] * v[b,n,h,e] ----------------
__global__ __launch_bounds__(256) void ctx_kernel(const float* __restrict__ k,
                                                  const float* __restrict__ v,
                                                  float* __restrict__ ctx) {
  int wave = threadIdx.x >> 5, lane = threadIdx.x & 31;
  int gw = blockIdx.x * 8 + wave;  // total BB*HH*DH = 4096 waves
  int d = gw % DH;
  int hh = (gw / DH) % HH;
  int b = gw / (DH * HH);
  const float* kp = k + (size_t)b * NT * DD + hh * DH + d;
  const float* vp = v + (size_t)b * NT * DD + hh * DH + lane;
  float acc = 0.f;
  for (int n = 0; n < NT; ++n)
    acc += kp[(size_t)n * DD] * vp[(size_t)n * DD];
  ctx[(((size_t)b * HH + hh) * DH + d) * DH + lane] = acc;
}

// ---------------- o[b,n,h,e] = sum_d q[b,n,h,d] * ctx[b,h,d,e] -> bf16 --------
__global__ __launch_bounds__(256) void o_kernel(const float* __restrict__ q,
                                                const float* __restrict__ ctx,
                                                bf16_t* __restrict__ o) {
  int wave = threadIdx.x >> 5, lane = threadIdx.x & 31;
  int gw = blockIdx.x * 8 + wave;  // total BB*NT*HH = 262272 waves (exact /8)
  if (gw >= BB * NT * HH) return;
  int hh = gw % HH;
  int n = (gw / HH) % NT;
  int b = gw / (HH * NT);
  const float* qp = q + ((size_t)b * NT + n) * DD + hh * DH;
  const float* cp = ctx + (((size_t)b * HH + hh) * DH) * DH;
  float acc = 0.f;
#pragma unroll
  for (int d = 0; d < DH; ++d) acc += qp[d] * cp[d * DH + lane];
  o[((size_t)b * NT + n) * DD + hh * DH + lane] = (bf16_t)acc;
}

// ---------------- classifier head: out[b,c] = h[b,0,:] @ Wh + bh --------------
__global__ void head_kernel(const float* __restrict__ h, const float* __restrict__ Wh,
                            const float* __restrict__ bh, float* __restrict__ out) {
  int t = threadIdx.x;
  if (t >= BB * NCLS) return;
  int b = t / NCLS, c = t % NCLS;
  const float* hp = h + (size_t)b * NT * DD;  // row n=0
  float s = bh[c];
  for (int d = 0; d < DD; ++d) s += hp[d] * Wh[d * NCLS + c];
  out[t] = s;
}

// ============================= host launcher =================================
extern "C" void kernel_launch(void* const* d_in, const int* in_sizes, int n_in,
                              void* d_out, int out_size, void* d_ws, size_t ws_size,
                              hipStream_t stream) {
  (void)in_sizes; (void)n_in; (void)out_size; (void)ws_size;
  const int*   x   = (const int*)  d_in[0];
  const float* emb = (const float*)d_in[1];
  const float* cht = (const float*)d_in[2];
  const float* cls = (const float*)d_in[3];
  const float* Wq  = (const float*)d_in[4];
  const float* Wk  = (const float*)d_in[5];
  const float* Wv  = (const float*)d_in[6];
  const float* Wo  = (const float*)d_in[7];
  const float* bo  = (const float*)d_in[8];
  const float* g1  = (const float*)d_in[9];
  const float* b1  = (const float*)d_in[10];
  const float* W1  = (const float*)d_in[11];
  const float* bf1 = (const float*)d_in[12];
  const float* W2  = (const float*)d_in[13];
  const float* bf2 = (const float*)d_in[14];
  const float* g2  = (const float*)d_in[15];
  const float* b2  = (const float*)d_in[16];
  const float* Wh  = (const float*)d_in[17];
  const float* bh  = (const float*)d_in[18];
  float* out = (float*)d_out;

  // ---- workspace carve (256B aligned slabs) ----
  char* ws = (char*)d_ws;
  auto carve = [&](size_t bytes) -> char* {
    char* p = ws;
    ws += (bytes + 255) & ~(size_t)255;
    return p;
  };
  float*  h    = (float*) carve((size_t)BB * NT * DD * 4);
  float*  qf   = (float*) carve((size_t)BB * NT * DD * 4);
  float*  kf   = (float*) carve((size_t)BB * NT * DD * 4);
  float*  vf   = (float*) carve((size_t)BB * NT * DD * 4);
  float*  ctx  = (float*) carve((size_t)BB * HH * DH * DH * 4);
  bf16_t* ybf  = (bf16_t*)carve((size_t)BB * NT * DD * 2);
  bf16_t* obf  = (bf16_t*)carve((size_t)BB * NT * DD * 2);
  bf16_t* f1b  = (bf16_t*)carve((size_t)BB * NT * FFD * 2);
  bf16_t* wqb  = (bf16_t*)carve((size_t)LL * DD * DD * 2);
  bf16_t* wkb  = (bf16_t*)carve((size_t)LL * DD * DD * 2);
  bf16_t* wvb  = (bf16_t*)carve((size_t)LL * DD * DD * 2);
  bf16_t* wob  = (bf16_t*)carve((size_t)LL * DD * DD * 2);
  bf16_t* w1b  = (bf16_t*)carve((size_t)LL * DD * FFD * 2);
  bf16_t* w2b  = (bf16_t*)carve((size_t)LL * FFD * DD * 2);

  // ---- one-shot weight conversion to bf16 (layout kept [K][N]) ----
  auto cvt = [&](const float* s, bf16_t* d, size_t n) {
    cvt_bf16_kernel<<<(unsigned)((n + 255) / 256), 256, 0, stream>>>(s, d, n);
  };
  cvt(Wq, wqb, (size_t)LL * DD * DD);
  cvt(Wk, wkb, (size_t)LL * DD * DD);
  cvt(Wv, wvb, (size_t)LL * DD * DD);
  cvt(Wo, wob, (size_t)LL * DD * DD);
  cvt(W1, w1b, (size_t)LL * DD * FFD);
  cvt(W2, w2b, (size_t)LL * FFD * DD);

  // ---- embedding ----
  embed_kernel<<<dim3(NT, BB), DD, 0, stream>>>(x, emb, cht, cls, h);

  const dim3 gD (DD / 64,  (NT + 63) / 64, BB);   // N=256 GEMMs
  const dim3 gFF(FFD / 64, (NT + 63) / 64, BB);   // N=1024 GEMM

  for (int l = 0; l < LL; ++l) {
    const bf16_t* wq = wqb + (size_t)l * DD * DD;
    const bf16_t* wk = wkb + (size_t)l * DD * DD;
    const bf16_t* wv = wvb + (size_t)l * DD * DD;
    const bf16_t* wo = wob + (size_t)l * DD * DD;
    const bf16_t* w1 = w1b + (size_t)l * DD * FFD;
    const bf16_t* w2 = w2b + (size_t)l * FFD * DD;

    // --- PreNorm linear attention with residual ---
    ln_kernel<<<dim3(NT, BB), DD, 0, stream>>>(h, g1 + l * DD, b1 + l * DD, ybf);
    gemm_ws_kernel<0, 0, 0, 0><<<gD, 256, 0, stream>>>(ybf, wq, nullptr, qf, NT, DD, DD);
    gemm_ws_kernel<0, 0, 0, 0><<<gD, 256, 0, stream>>>(ybf, wk, nullptr, kf, NT, DD, DD);
    gemm_ws_kernel<0, 0, 0, 0><<<gD, 256, 0, stream>>>(ybf, wv, nullptr, vf, NT, DD, DD);
    qsoftmax_kernel<<<(BB * NT * HH + 255) / 256, 256, 0, stream>>>(qf);
    ksoftmax_kernel<<<dim3(DD, BB), 256, 0, stream>>>(kf);
    ctx_kernel<<<BB * HH * DH / 8, 256, 0, stream>>>(kf, vf, ctx);
    o_kernel<<<BB * NT * HH / 8, 256, 0, stream>>>(qf, ctx, obf);
    gemm_ws_kernel<1, 1, 0, 0><<<gD, 256, 0, stream>>>(obf, wo, bo + l * DD, h, NT, DD, DD);

    // --- PreNorm feed-forward with residual ---
    ln_kernel<<<dim3(NT, BB), DD, 0, stream>>>(h, g2 + l * DD, b2 + l * DD, ybf);
    gemm_ws_kernel<1, 0, 1, 1><<<gFF, 256, 0, stream>>>(ybf, w1, bf1 + l * FFD, f1b, NT, FFD, DD);
    gemm_ws_kernel<1, 1, 0, 0><<<gD, 256, 0, stream>>>(f1b, w2, bf2 + l * DD, h, NT, DD, FFD);
  }

  head_kernel<<<1, 128, 0, stream>>>(h, Wh, bh, out);
}
